// UnfeatGraphConvNet_24154896073101
// MI455X (gfx1250) — matrix operand order repaired and checked by reading, not compiled, not run on hardware
//
#include <hip/hip_runtime.h>
#include <hip/hip_bf16.h>
#include <math.h>

typedef _Float16 h16;
typedef __attribute__((ext_vector_type(16))) _Float16 v16h;
typedef __attribute__((ext_vector_type(8)))  _Float16 v8h;
typedef __attribute__((ext_vector_type(8)))  float    v8f;

#define NN 100000
#define EE 1600000
#define HD 128
#define CO 40
#define CP 48
#define ROWTILES (NN / 16)   // 6250, exact
#define ROWS_PER_BLK 64

__device__ __forceinline__ void atomAddF(float* p, float v) {
  __hip_atomic_fetch_add(p, v, __ATOMIC_RELAXED, __HIP_MEMORY_SCOPE_AGENT);
}

// ---------------- utility kernels ----------------
__global__ void k_zero(float* __restrict__ p, long n) {
  long i = (long)blockIdx.x * blockDim.x + threadIdx.x;
  if (i < n) p[i] = 0.f;
}

__global__ void k_degree(const int* __restrict__ src, const int* __restrict__ dst,
                         float* __restrict__ dout, float* __restrict__ din) {
  int e = blockIdx.x * blockDim.x + threadIdx.x;
  if (e < EE) {
    atomAddF(&dout[src[e]], 1.f);
    atomAddF(&din[dst[e]], 1.f);
  }
}

__global__ void k_invsqrt(float* __restrict__ d, int n) {
  int i = blockIdx.x * blockDim.x + threadIdx.x;
  if (i < n) d[i] = rsqrtf(fmaxf(d[i], 1.f));
}

__global__ void k_f32_to_f16(const float* __restrict__ in, h16* __restrict__ out, long n) {
  long i = (long)blockIdx.x * blockDim.x + threadIdx.x;
  if (i < n) out[i] = (h16)in[i];
}

// Wt[n*128 + k] = W[k*outc + n], zero-padded to cpad output columns.
__global__ void k_transpose_w(const float* __restrict__ W, h16* __restrict__ Wt,
                              int outc, int cpad) {
  int i = blockIdx.x * blockDim.x + threadIdx.x;
  if (i < cpad * HD) {
    int n = i / HD, k = i % HD;
    float v = (n < outc) ? W[k * outc + n] : 0.f;
    Wt[i] = (h16)v;
  }
}

// ---------------- WMMA GEMMs ----------------
// Y[m][n] = (sum_k A[m][k] * Wt[n][k]) * rowScale[m] ; A: [rows x 128] f16, Wt: [128 x 128] f16
__global__ void k_gemm128_scale(const h16* __restrict__ A, const h16* __restrict__ Bt,
                                const float* __restrict__ rowScale, float* __restrict__ Y) {
  const int wave = threadIdx.x >> 5;          // 8 waves -> 8 col tiles of 16
  const int lane = threadIdx.x & 31;
  const int m  = blockIdx.x * 16 + (lane & 15);
  const int n  = wave * 16 + (lane & 15);
  const int ka = (lane >> 4) * 8;             // A: halfs 0..7 -> K=ka.., 8..15 -> K=16+ka..
  const int kb = (lane >> 4) * 8;             // B: halfs 0..7 -> K=kb.., 8..15 -> K=16+kb..
  const h16* Ar = A  + (size_t)m * HD;
  const h16* Br = Bt + (size_t)n * HD;
  v8f acc = {};
#pragma unroll
  for (int kk = 0; kk < HD; kk += 32) {
    v8h a0 = *(const v8h*)(Ar + kk + ka);
    v8h a1 = *(const v8h*)(Ar + kk + 16 + ka);
    v8h b0 = *(const v8h*)(Br + kk + kb);
    v8h b1 = *(const v8h*)(Br + kk + 16 + kb);
    v16h a = __builtin_shufflevector(a0, a1, 0,1,2,3,4,5,6,7,8,9,10,11,12,13,14,15);
    v16h b = __builtin_shufflevector(b0, b1, 0,1,2,3,4,5,6,7,8,9,10,11,12,13,14,15);
    acc = __builtin_amdgcn_wmma_f32_16x16x32_f16(false, a, false, b, (short)0, acc, false, false);
  }
  const int mrb = blockIdx.x * 16 + ((lane >> 4) << 3);
#pragma unroll
  for (int r = 0; r < 8; ++r) {
    int mr = mrb + r;
    Y[(size_t)mr * HD + n] = acc[r] * rowScale[mr];
  }
}

// out[m][c] = feat16[m][:] @ Wout + bout ; Bt: [48 x 128] padded; writes c < 40 only.
__global__ void k_gemm_out(const h16* __restrict__ A, const h16* __restrict__ Bt,
                           const float* __restrict__ bias, float* __restrict__ out) {
  const int wave = threadIdx.x >> 5;          // 3 waves -> 48 padded cols
  const int lane = threadIdx.x & 31;
  const int m  = blockIdx.x * 16 + (lane & 15);
  const int n  = wave * 16 + (lane & 15);
  const int ka = (lane >> 4) * 8;
  const int kb = (lane >> 4) * 8;
  const h16* Ar = A  + (size_t)m * HD;
  const h16* Br = Bt + (size_t)n * HD;
  v8f acc = {};
#pragma unroll
  for (int kk = 0; kk < HD; kk += 32) {
    v8h a0 = *(const v8h*)(Ar + kk + ka);
    v8h a1 = *(const v8h*)(Ar + kk + 16 + ka);
    v8h b0 = *(const v8h*)(Br + kk + kb);
    v8h b1 = *(const v8h*)(Br + kk + 16 + kb);
    v16h a = __builtin_shufflevector(a0, a1, 0,1,2,3,4,5,6,7,8,9,10,11,12,13,14,15);
    v16h b = __builtin_shufflevector(b0, b1, 0,1,2,3,4,5,6,7,8,9,10,11,12,13,14,15);
    acc = __builtin_amdgcn_wmma_f32_16x16x32_f16(false, a, false, b, (short)0, acc, false, false);
  }
  if (n < CO) {
    const float bb = bias[n];
    const int mrb = blockIdx.x * 16 + ((lane >> 4) << 3);
#pragma unroll
    for (int r = 0; r < 8; ++r)
      out[(size_t)(mrb + r) * CO + n] = acc[r] + bb;
  }
}

// ---------------- SpMM scatter: one wave per edge ----------------
__global__ void k_scatter(const float* __restrict__ hsrc, const int* __restrict__ src,
                          const int* __restrict__ dst, float* __restrict__ agg) {
  long t = (long)blockIdx.x * blockDim.x + threadIdx.x;
  long e = t >> 5;
  if (e < EE) {
    int j = ((int)t & 31) * 4;
    int s = src[e], d = dst[e];
    const float4 v = *(const float4*)(hsrc + (size_t)s * HD + j);
    float* p = agg + (size_t)d * HD + j;
    atomAddF(p + 0, v.x); atomAddF(p + 1, v.y);
    atomAddF(p + 2, v.z); atomAddF(p + 3, v.w);
  }
}

// hpre = agg*inv_d + b written in place; per-column partial sums for BN.
__global__ void k_stats(float* __restrict__ agg, const float* __restrict__ invd,
                        const float* __restrict__ bias, float* __restrict__ gsum,
                        float* __restrict__ gsq) {
  const int c = threadIdx.x;                  // 128 threads = 128 columns
  const int r0 = blockIdx.x * ROWS_PER_BLK;
  const float b = bias[c];
  float s = 0.f, s2 = 0.f;
  for (int i = 0; i < ROWS_PER_BLK; ++i) {
    int r = r0 + i;
    if (r < NN) {
      size_t idx = (size_t)r * HD + c;
      float hp = agg[idx] * invd[r] + b;
      agg[idx] = hp;
      s += hp; s2 += hp * hp;
    }
  }
  atomAddF(&gsum[c], s);
  atomAddF(&gsq[c], s2);
}

__global__ void k_bn_finalize(const float* __restrict__ gsum, const float* __restrict__ gsq,
                              const float* __restrict__ g, const float* __restrict__ be,
                              float* __restrict__ scale, float* __restrict__ shift) {
  int c = threadIdx.x;
  float m = gsum[c] / (float)NN;
  float v = gsq[c] / (float)NN - m * m;
  float sc = g[c] * rsqrtf(v + 1e-5f);
  scale[c] = sc;
  shift[c] = be[c] - m * sc;
}

__global__ void k_bn_relu(float* __restrict__ hpre, const float* __restrict__ scale,
                          const float* __restrict__ shift, h16* __restrict__ out16) {
  long i = (long)blockIdx.x * blockDim.x + threadIdx.x;
  if (i < (long)NN * HD) {
    int c = (int)(i & (HD - 1));
    float h = fmaxf(hpre[i] * scale[c] + shift[c], 0.f);
    hpre[i] = h;
    if (out16) out16[i] = (h16)h;
  }
}

// Row-wise L2 normalize: one wave (32 lanes x float4) per row.
__global__ void k_l2norm(const float* __restrict__ h, float* __restrict__ feat,
                         h16* __restrict__ feat16) {
  int wave = threadIdx.x >> 5, lane = threadIdx.x & 31;
  int row = blockIdx.x * 8 + wave;
  if (row >= NN) return;
  const float4 v = *(const float4*)(h + (size_t)row * HD + lane * 4);
  float s = v.x * v.x + v.y * v.y + v.z * v.z + v.w * v.w;
  for (int off = 16; off > 0; off >>= 1) s += __shfl_xor(s, off, 32);
  float inv = 1.f / fmaxf(sqrtf(s), 1e-12f);
  float4 o; o.x = v.x * inv; o.y = v.y * inv; o.z = v.z * inv; o.w = v.w * inv;
  *(float4*)(feat + (size_t)row * HD + lane * 4) = o;
  h16* p16 = feat16 + (size_t)row * HD + lane * 4;
  p16[0] = (h16)o.x; p16[1] = (h16)o.y; p16[2] = (h16)o.z; p16[3] = (h16)o.w;
}

// ---------------- launcher ----------------
extern "C" void kernel_launch(void* const* d_in, const int* in_sizes, int n_in,
                              void* d_out, int out_size, void* d_ws, size_t ws_size,
                              hipStream_t stream) {
  (void)in_sizes; (void)n_in; (void)out_size; (void)ws_size;
  const float* x    = (const float*)d_in[0];
  const int*   esrc = (const int*)d_in[1];
  const int*   edst = (const int*)d_in[2];
  const float* W1   = (const float*)d_in[3];
  const float* b1   = (const float*)d_in[4];
  const float* g1   = (const float*)d_in[5];
  const float* be1  = (const float*)d_in[6];
  const float* W2   = (const float*)d_in[7];
  const float* b2   = (const float*)d_in[8];
  const float* g2   = (const float*)d_in[9];
  const float* be2  = (const float*)d_in[10];
  const float* Wout = (const float*)d_in[11];
  const float* bout = (const float*)d_in[12];

  float* out  = (float*)d_out;              // [N, 40]
  float* feat = out + (size_t)NN * CO;      // [N, 128]

  char* ws = (char*)d_ws;
  size_t o = 0;
  auto alloc = [&](size_t bytes) -> void* {
    void* p = ws + o;
    o += (bytes + 255) & ~(size_t)255;
    return p;
  };
  h16*   x16  = (h16*)alloc((size_t)NN * HD * 2);   // layer-1 input f16; reused as feat16
  h16*   h1_16= (h16*)alloc((size_t)NN * HD * 2);   // layer-1 output f16 (GEMM2 input)
  float* hs   = (float*)alloc((size_t)NN * HD * 4); // scaled GEMM output
  float* agg  = (float*)alloc((size_t)NN * HD * 4); // agg -> hpre -> h (in place)
  float* invs = (float*)alloc((size_t)NN * 4);
  float* invd = (float*)alloc((size_t)NN * 4);
  h16*   W1t  = (h16*)alloc(HD * HD * 2);
  h16*   W2t  = (h16*)alloc(HD * HD * 2);
  h16*   Wot  = (h16*)alloc(CP * HD * 2);
  float* stats = (float*)alloc(512 * 4);
  float *gsum = stats, *gsq = stats + 128, *scale = stats + 256, *shift = stats + 384;

  const long NH = (long)NN * HD;
  const int BLK = 256;
  dim3 gNH((unsigned)((NH + BLK - 1) / BLK));
  dim3 gN((unsigned)((NN + BLK - 1) / BLK));
  dim3 gE((unsigned)((EE + BLK - 1) / BLK));
  dim3 gScat((unsigned)(((long)EE * 32 + BLK - 1) / BLK));
  dim3 gStats((unsigned)((NN + ROWS_PER_BLK - 1) / ROWS_PER_BLK));

  // degrees -> inv_s, inv_d
  k_zero<<<gN, BLK, 0, stream>>>(invs, NN);
  k_zero<<<gN, BLK, 0, stream>>>(invd, NN);
  k_degree<<<gE, BLK, 0, stream>>>(esrc, edst, invs, invd);
  k_invsqrt<<<gN, BLK, 0, stream>>>(invs, NN);
  k_invsqrt<<<gN, BLK, 0, stream>>>(invd, NN);

  // precision conversion / weight transpose
  k_f32_to_f16<<<gNH, BLK, 0, stream>>>(x, x16, NH);
  k_transpose_w<<<dim3((HD * HD + BLK - 1) / BLK), BLK, 0, stream>>>(W1, W1t, HD, HD);
  k_transpose_w<<<dim3((HD * HD + BLK - 1) / BLK), BLK, 0, stream>>>(W2, W2t, HD, HD);
  k_transpose_w<<<dim3((CP * HD + BLK - 1) / BLK), BLK, 0, stream>>>(Wout, Wot, CO, CP);

  // ---- layer 1 ----
  k_gemm128_scale<<<dim3(ROWTILES), 256, 0, stream>>>(x16, W1t, invs, hs);
  k_zero<<<gNH, BLK, 0, stream>>>(agg, NH);
  k_zero<<<dim3(1), 256, 0, stream>>>(gsum, 256);  // gsum + gsq contiguous
  k_scatter<<<gScat, BLK, 0, stream>>>(hs, esrc, edst, agg);
  k_stats<<<gStats, HD, 0, stream>>>(agg, invd, b1, gsum, gsq);
  k_bn_finalize<<<dim3(1), HD, 0, stream>>>(gsum, gsq, g1, be1, scale, shift);
  k_bn_relu<<<gNH, BLK, 0, stream>>>(agg, scale, shift, h1_16);

  // ---- layer 2 ----
  k_gemm128_scale<<<dim3(ROWTILES), 256, 0, stream>>>(h1_16, W2t, invs, hs);
  k_zero<<<gNH, BLK, 0, stream>>>(agg, NH);
  k_zero<<<dim3(1), 256, 0, stream>>>(gsum, 256);
  k_scatter<<<gScat, BLK, 0, stream>>>(hs, esrc, edst, agg);
  k_stats<<<gStats, HD, 0, stream>>>(agg, invd, b2, gsum, gsq);
  k_bn_finalize<<<dim3(1), HD, 0, stream>>>(gsum, gsq, g2, be2, scale, shift);
  k_bn_relu<<<gNH, BLK, 0, stream>>>(agg, scale, shift, (h16*)nullptr);

  // ---- normalize + output GEMM ----
  k_l2norm<<<dim3((NN + 7) / 8), 256, 0, stream>>>(agg, feat, x16 /*feat16, x dead*/);
  k_gemm_out<<<dim3(ROWTILES), 96, 0, stream>>>(x16, Wot, bout, out);
}
